// AttentionBlock_51342039056796
// MI455X (gfx1250) — matrix operand order repaired and checked by reading, not compiled, not run on hardware
//
#include <hip/hip_runtime.h>
#include <hip/hip_bf16.h>

// ---------------------------------------------------------------------------
// AttentionBlock for MI455X (gfx1250), wave32 + WMMA f16.
// All GEMMs use v_wmma_f32_16x16x32_f16 (f32 accumulate). Softmax tile is
// LDS-resident (96 KB/block < 320 KB WGP LDS). f32 kept only for accumulators,
// softmax, bias/Swish epilogue and the residual.
// ---------------------------------------------------------------------------

typedef __attribute__((ext_vector_type(16))) _Float16 v16h;
typedef __attribute__((ext_vector_type(8)))  _Float16 v8h;
typedef __attribute__((ext_vector_type(8)))  float    v8f;

#define S_TOK 1024   // H*W tokens
#define C_DIM 256    // channels
#define HD    256    // heads * d_k
#define NH    4
#define DK    64
#define B_N   16

// A operand 16x32 f16 (MxK), row-major source with leading dim `ld` halves.
// Lane L -> row L&15; halves [0..15] = contiguous K 16*(L>=16)+0..15
// (matches load_b_32x16).
__device__ __forceinline__ v16h load_a_16x32(const _Float16* __restrict__ base,
                                             int ld, int lane, int col) {
  const int r = lane & 15, hs = lane >> 4;
  const _Float16* row = base + (size_t)r * ld + col;
  v8h c0 = *(const v8h*)(row + hs * 16);
  v8h c1 = *(const v8h*)(row + hs * 16 + 8);
  v16h o;
#pragma unroll
  for (int i = 0; i < 8; ++i) { o[i] = c0[i]; o[i + 8] = c1[i]; }
  return o;
}

// B operand 32x16 f16 (KxN) supplied as B^T rows: baseT is row-major [N x K],
// leading dim `ld` halves. Lane L -> column n=L&15, contiguous K chunk
// 16*(L>=16)+0..15 (same k order as load_a_16x32).
__device__ __forceinline__ v16h load_b_32x16(const _Float16* __restrict__ baseT,
                                             int ld, int lane, int col) {
  const int n = lane & 15, hs = lane >> 4;
  return *(const v16h*)(baseT + (size_t)n * ld + col + hs * 16);
}

__device__ __forceinline__ v8f wmma_f16(v16h a, v16h b, v8f c) {
  return __builtin_amdgcn_wmma_f32_16x16x32_f16(false, a, false, b,
                                                (short)0, c, false, false);
}

// ---------------------------------------------------------------------------
// Kernel 0: convert weights f32 -> f16 once per launch.
// ---------------------------------------------------------------------------
__global__ void __launch_bounds__(256)
cvt_weights_kernel(const float* __restrict__ wp, const float* __restrict__ wo,
                   _Float16* __restrict__ wp_h, _Float16* __restrict__ wo_h) {
  const int i = blockIdx.x * 256 + threadIdx.x;
  if (i < 3 * HD * C_DIM) wp_h[i] = (_Float16)wp[i];
  if (i < C_DIM * HD)     wo_h[i] = (_Float16)wo[i];
}

// ---------------------------------------------------------------------------
// Kernel 1: GroupNorm over (8 ch, 32, 32) per (b,g); emits tokens xs[b,s,c]
// in f32 (residual path) and f16 (GEMM A operand).
// ---------------------------------------------------------------------------
__global__ void __launch_bounds__(256)
groupnorm_kernel(const float* __restrict__ x,
                 const float* __restrict__ gn_w, const float* __restrict__ gn_b,
                 float* __restrict__ xs_f32, _Float16* __restrict__ xs_h) {
  const int CPG = 8;                      // channels per group (256/32)
  const int b = blockIdx.x >> 5;
  const int g = blockIdx.x & 31;
  const int c0 = g * CPG;
  const float* xb = x + ((size_t)b * C_DIM + c0) * S_TOK;

  float s = 0.f, ss = 0.f;
  for (int e = threadIdx.x; e < CPG * S_TOK; e += 256) {
    const float v = xb[(e & 7) * S_TOK + (e >> 3)];
    s += v; ss += v * v;
  }
  __shared__ float rs[256], rss[256];
  rs[threadIdx.x] = s; rss[threadIdx.x] = ss;
  __syncthreads();
  for (int o = 128; o > 0; o >>= 1) {
    if (threadIdx.x < (unsigned)o) {
      rs[threadIdx.x] += rs[threadIdx.x + o];
      rss[threadIdx.x] += rss[threadIdx.x + o];
    }
    __syncthreads();
  }
  __shared__ float mean_s, rstd_s;
  if (threadIdx.x == 0) {
    const float inv_n = 1.0f / (CPG * S_TOK);
    const float m = rs[0] * inv_n;
    const float var = rss[0] * inv_n - m * m;
    mean_s = m;
    rstd_s = rsqrtf(var + 1e-5f);
  }
  __syncthreads();
  const float m = mean_s, r = rstd_s;
  for (int e = threadIdx.x; e < CPG * S_TOK; e += 256) {
    const int ci = e & 7, p = e >> 3, c = c0 + ci;
    const float v = (xb[ci * S_TOK + p] - m) * r * gn_w[c] + gn_b[c];
    const size_t o = ((size_t)b * S_TOK + p) * C_DIM + c;
    xs_f32[o] = v;
    xs_h[o] = (_Float16)v;
  }
}

// ---------------------------------------------------------------------------
// Kernel 2: QKV GEMM  [1024,256] @ w_proj^T[256,768] + bias, scattered into
// Q[b,h,s,dk], K[b,h,s,dk], V^T[b,h,dk,s] (all f16) for the attention GEMMs.
// 256 threads = 8 waves; wave -> (i_sub = w&1, j-tile group = w>>1).
// ---------------------------------------------------------------------------
__global__ void __launch_bounds__(256)
qkv_kernel(const _Float16* __restrict__ Xh, const _Float16* __restrict__ Wp,
           const float* __restrict__ bp,
           _Float16* __restrict__ Qh, _Float16* __restrict__ Kh,
           _Float16* __restrict__ Vt) {
  const int blk = blockIdx.x;
  const int st = blk & 31;          // 32-row super-tile
  const int b  = blk >> 5;
  const int wv = threadIdx.x >> 5, lane = threadIdx.x & 31;
  const int isub = wv & 1, grp = wv >> 1;
  const int s0 = st * 32 + isub * 16;
  const _Float16* Ab = Xh + ((size_t)b * S_TOK + s0) * C_DIM;

  // Hoist the 8 A fragments (K = 256) for this 16-row stripe.
  v16h afrag[8];
#pragma unroll
  for (int kk = 0; kk < 8; ++kk) afrag[kk] = load_a_16x32(Ab, C_DIM, lane, kk * 32);

  const int n = lane & 15, hs = lane >> 4;
  for (int t = 0; t < 12; ++t) {
    const int jt = grp + 4 * t;     // 48 j-tiles over 4 wave-groups
    v8f acc = {};
#pragma unroll
    for (int kk = 0; kk < 8; ++kk) {
      v16h bm = load_b_32x16(Wp + (size_t)jt * 16 * C_DIM, C_DIM, lane, kk * 32);
      acc = wmma_f16(afrag[kk], bm, acc);
    }
    const int j = jt * 16 + n;      // row of w_proj in [0, 768)
    const int head = j / 192, rr = j % 192;
    const float bias = bp[j];
    const size_t bh = (size_t)b * NH + head;
#pragma unroll
    for (int v = 0; v < 8; ++v) {
      const int s = s0 + v + 8 * hs;
      const _Float16 val = (_Float16)(acc[v] + bias);
      if (rr < 64)        Qh[(bh * S_TOK + s) * DK + rr] = val;
      else if (rr < 128)  Kh[(bh * S_TOK + s) * DK + (rr - 64)] = val;
      else                Vt[(bh * DK + (rr - 128)) * S_TOK + s] = val;
    }
  }
}

// ---------------------------------------------------------------------------
// Kernel 3: fused attention per (b, h, 16-row i-tile).
//   scores = Q_tile @ K^T * sqrt(64)  (WMMA, f32 LDS tile 16x1024 = 64 KB)
//   softmax over j in LDS, probs -> f16 (32 KB)
//   out = probs @ V (WMMA over K=1024 using V^T rows), f16 to Ho[b,s,h*64+d]
// 128 threads = 4 waves.
// ---------------------------------------------------------------------------
__global__ void __launch_bounds__(128)
attn_kernel(const _Float16* __restrict__ Qh, const _Float16* __restrict__ Kh,
            const _Float16* __restrict__ Vt, _Float16* __restrict__ Ho) {
  __shared__ float    sc[16][S_TOK];   // 64 KB
  __shared__ _Float16 pr[16][S_TOK];   // 32 KB
  const int blk = blockIdx.x;
  const int st = blk & 63;
  const int h  = (blk >> 6) & 3;
  const int b  = blk >> 8;
  const int s0 = st * 16;
  const int wv = threadIdx.x >> 5, lane = threadIdx.x & 31;
  const size_t bh = (size_t)b * NH + h;
  const _Float16* Qb = Qh + (bh * S_TOK + s0) * DK;
  const _Float16* Kb = Kh + bh * S_TOK * DK;
  const _Float16* Vb = Vt + bh * (size_t)DK * S_TOK;
  const int n = lane & 15, hs = lane >> 4;

  // ---- phase 1: scores ----
  v16h qf[2];
#pragma unroll
  for (int kk = 0; kk < 2; ++kk) qf[kk] = load_a_16x32(Qb, DK, lane, kk * 32);
  for (int t = 0; t < 16; ++t) {
    const int jt = wv + 4 * t;        // 64 j-tiles over 4 waves
    v8f acc = {};
#pragma unroll
    for (int kk = 0; kk < 2; ++kk) {
      v16h bm = load_b_32x16(Kb + (size_t)jt * 16 * DK, DK, lane, kk * 32);
      acc = wmma_f16(qf[kk], bm, acc);
    }
#pragma unroll
    for (int v = 0; v < 8; ++v)
      sc[v + 8 * hs][jt * 16 + n] = acc[v] * 8.0f;   // * sqrt(d_k)
  }
  __syncthreads();

  // ---- phase 2: softmax over j (4 rows per wave) ----
  for (int k = 0; k < 4; ++k) {
    const int ri = wv * 4 + k;
    float mx = -3.4e38f;
    for (int j = lane; j < S_TOK; j += 32) mx = fmaxf(mx, sc[ri][j]);
#pragma unroll
    for (int o = 16; o > 0; o >>= 1) mx = fmaxf(mx, __shfl_xor(mx, o, 32));
    float sum = 0.f;
    for (int j = lane; j < S_TOK; j += 32) {
      const float e = __expf(sc[ri][j] - mx);
      sc[ri][j] = e;
      sum += e;
    }
#pragma unroll
    for (int o = 16; o > 0; o >>= 1) sum += __shfl_xor(sum, o, 32);
    const float inv = 1.0f / sum;
    for (int j = lane; j < S_TOK; j += 32) pr[ri][j] = (_Float16)(sc[ri][j] * inv);
  }
  __syncthreads();

  // ---- phase 3: out = probs @ V, one 16x16 d-tile per wave ----
  {
    const int d0 = wv * 16;
    v8f acc = {};
    for (int jc = 0; jc < S_TOK; jc += 32) {
      v16h a  = load_a_16x32(&pr[0][0], S_TOK, lane, jc);
      v16h bm = load_b_32x16(Vb + (size_t)d0 * S_TOK, S_TOK, lane, jc);
      acc = wmma_f16(a, bm, acc);
    }
#pragma unroll
    for (int v = 0; v < 8; ++v) {
      const int m = v + 8 * hs;
      Ho[((size_t)b * S_TOK + s0 + m) * HD + h * DK + d0 + n] = (_Float16)acc[v];
    }
  }
}

// ---------------------------------------------------------------------------
// Kernel 4: out-projection + bias + Swish + residual(xs_f32) + NCHW store.
// 128 threads = 4 waves; 16-row s-tile, 16 channel-tiles over 4 waves.
// ---------------------------------------------------------------------------
__global__ void __launch_bounds__(128)
proj_kernel(const _Float16* __restrict__ Ho, const _Float16* __restrict__ Wo,
            const float* __restrict__ bo, const float* __restrict__ xs,
            float* __restrict__ out) {
  const int blk = blockIdx.x;
  const int st = blk & 63;
  const int b  = blk >> 6;
  const int s0 = st * 16;
  const int wv = threadIdx.x >> 5, lane = threadIdx.x & 31;
  const _Float16* Ab = Ho + ((size_t)b * S_TOK + s0) * HD;

  v16h afrag[8];
#pragma unroll
  for (int kk = 0; kk < 8; ++kk) afrag[kk] = load_a_16x32(Ab, HD, lane, kk * 32);

  const int n = lane & 15, hs = lane >> 4;
  for (int t = 0; t < 4; ++t) {
    const int nt = wv + 4 * t;        // 16 channel-tiles over 4 waves
    v8f acc = {};
#pragma unroll
    for (int kk = 0; kk < 8; ++kk) {
      v16h bm = load_b_32x16(Wo + (size_t)nt * 16 * HD, HD, lane, kk * 32);
      acc = wmma_f16(afrag[kk], bm, acc);
    }
    const int c = nt * 16 + n;
    const float bias = bo[c];
#pragma unroll
    for (int v = 0; v < 8; ++v) {
      const int s = s0 + v + 8 * hs;
      float val = acc[v] + bias;
      val = val / (1.0f + __expf(-val));                       // Swish
      val += xs[((size_t)b * S_TOK + s) * C_DIM + c];          // residual
      out[((size_t)b * C_DIM + c) * S_TOK + s] = val;          // NCHW
    }
  }
}

// ---------------------------------------------------------------------------
extern "C" void kernel_launch(void* const* d_in, const int* in_sizes, int n_in,
                              void* d_out, int out_size, void* d_ws, size_t ws_size,
                              hipStream_t stream) {
  const float* x      = (const float*)d_in[0];
  const float* w_proj = (const float*)d_in[1];
  const float* b_proj = (const float*)d_in[2];
  const float* gn_w   = (const float*)d_in[3];
  const float* gn_b   = (const float*)d_in[4];
  const float* w_out  = (const float*)d_in[5];
  const float* b_out  = (const float*)d_in[6];
  float* out = (float*)d_out;

  // Workspace carve-up (~59.5 MB total).
  char* ws = (char*)d_ws;
  float*    xs_f32 = (float*)ws;    ws += (size_t)B_N * S_TOK * C_DIM * 4;
  _Float16* xs_h   = (_Float16*)ws; ws += (size_t)B_N * S_TOK * C_DIM * 2;
  _Float16* wp_h   = (_Float16*)ws; ws += (size_t)3 * HD * C_DIM * 2;
  _Float16* wo_h   = (_Float16*)ws; ws += (size_t)C_DIM * HD * 2;
  _Float16* Qh     = (_Float16*)ws; ws += (size_t)B_N * NH * S_TOK * DK * 2;
  _Float16* Kh     = (_Float16*)ws; ws += (size_t)B_N * NH * S_TOK * DK * 2;
  _Float16* Vt     = (_Float16*)ws; ws += (size_t)B_N * NH * DK * S_TOK * 2;
  _Float16* Ho     = (_Float16*)ws; ws += (size_t)B_N * S_TOK * HD * 2;

  cvt_weights_kernel<<<(3 * HD * C_DIM) / 256, 256, 0, stream>>>(w_proj, w_out, wp_h, wo_h);
  groupnorm_kernel<<<B_N * 32, 256, 0, stream>>>(x, gn_w, gn_b, xs_f32, xs_h);
  qkv_kernel<<<B_N * (S_TOK / 32), 256, 0, stream>>>(xs_h, wp_h, b_proj, Qh, Kh, Vt);
  attn_kernel<<<B_N * NH * (S_TOK / 16), 128, 0, stream>>>(Qh, Kh, Vt, Ho);
  proj_kernel<<<B_N * (S_TOK / 16), 128, 0, stream>>>(Ho, wo_h, b_out, xs_f32, out);
}
